// MultiHeadAttention_29618094473469
// MI455X (gfx1250) — matrix operand order repaired and checked by reading, not compiled, not run on hardware
//
#include <hip/hip_runtime.h>
#include <hip/hip_bf16.h>

typedef __attribute__((ext_vector_type(16))) __bf16 v16bf;
typedef __attribute__((ext_vector_type(8)))  float  v8f;

#define B_   8
#define S_   1024
#define DM_  1024
#define H_   16
#define DK_  64

__device__ inline v8f wmma_bf16(v16bf a, v16bf b, v8f c) {
  return __builtin_amdgcn_wmma_f32_16x16x32_bf16(false, a, false, b, (short)0, c,
                                                 false, false);
}

// A-fragment (16 rows x 32 K, bf16) from a row-major matrix, leading dim ld.
__device__ inline v16bf load_frag_a(const __bf16* base, int ld, int r0, int kc0, int lane) {
  int row = r0 + (lane & 15);
  const __bf16* p = base + row * ld + kc0 + ((lane & 16) ? 8 : 0);
  union { uint4 q[2]; v16bf v; } u;
  u.q[0] = *reinterpret_cast<const uint4*>(p);
  u.q[1] = *reinterpret_cast<const uint4*>(p + 16);
  return u.v;
}

// B-fragment (32 K x 16 cols) from an N-major (B-transposed) matrix, ld = K stride.
__device__ inline v16bf load_frag_b(const __bf16* base, int ld, int n0, int kc0, int lane) {
  int row = n0 + (lane & 15);
  const __bf16* p = base + row * ld + kc0 + ((lane & 16) ? 8 : 0);
  union { uint4 q[2]; v16bf v; } u;
  u.q[0] = *reinterpret_cast<const uint4*>(p);
  u.q[1] = *reinterpret_cast<const uint4*>(p + 16);
  return u.v;
}

// ---------------------------------------------------------------- prep weights
__global__ __launch_bounds__(256) void mha_prep_weights(
    const float* __restrict__ Wq, const float* __restrict__ Wk,
    const float* __restrict__ Wv, const float* __restrict__ Wo,
    __bf16* __restrict__ WtQ, __bf16* __restrict__ WtK,
    __bf16* __restrict__ WtV, __bf16* __restrict__ WoB) {
  int i = blockIdx.x * blockDim.x + threadIdx.x;
  if (i >= DM_ * DM_) return;
  int h = i >> 16;
  int rem = i & 65535;
  int kk = rem >> 6;
  int d = rem & 63;
  int n = (h << 6) + d;
  size_t o = (size_t)n * DM_ + kk;
  WtQ[o] = (__bf16)Wq[i];
  WtK[o] = (__bf16)Wk[i];
  WtV[o] = (__bf16)Wv[i];
  WoB[i] = (__bf16)Wo[i];
}

// ---------------------------------------------------------------- projection GEMM
// A: [8192 x 1024] f32, Bt: [1024 x 1024] bf16 N-major, Out: bf16 [h][b][s][dk].
// 256 threads (8 waves), WG tile 128x128, wave tile 32x64, double-buffered LDS.
__global__ __launch_bounds__(256) void mha_proj_gemm(
    const float* __restrict__ A, const __bf16* __restrict__ Bt,
    __bf16* __restrict__ Out) {
  __shared__ __align__(16) __bf16 lA[2][128 * 32];
  __shared__ __align__(16) __bf16 lB[2][128 * 32];
  int tid = threadIdx.x, lane = tid & 31, w = tid >> 5;
  int m0 = blockIdx.x * 128, n0b = blockIdx.y * 128;
  int wm = w >> 1, wn = w & 1;  // wm: 0..3 (32 rows each), wn: 0..1 (64 cols each)
  v8f acc[2][4] = {};

  int srow = tid >> 1;             // 0..127
  int koff = (tid & 1) * 16;       // 0 or 16
  float4 ra[4];
  uint4 rb[2];

  auto load_regs = [&](int kb) {
    const float4* srcA = reinterpret_cast<const float4*>(A + (size_t)(m0 + srow) * DM_ + kb + koff);
    ra[0] = srcA[0]; ra[1] = srcA[1]; ra[2] = srcA[2]; ra[3] = srcA[3];
    const uint4* srcB = reinterpret_cast<const uint4*>(Bt + (size_t)(n0b + srow) * DM_ + kb + koff);
    rb[0] = srcB[0]; rb[1] = srcB[1];
    if (kb + 32 < DM_) {
      __builtin_prefetch(srcA + 8, 0, 1);
      __builtin_prefetch(srcB + 4, 0, 1);
    }
  };
  auto store_regs = [&](int buf) {
    float fbuf[16];
    reinterpret_cast<float4*>(fbuf)[0] = ra[0];
    reinterpret_cast<float4*>(fbuf)[1] = ra[1];
    reinterpret_cast<float4*>(fbuf)[2] = ra[2];
    reinterpret_cast<float4*>(fbuf)[3] = ra[3];
    union { __bf16 h[16]; uint4 q[2]; } u;
#pragma unroll
    for (int j = 0; j < 16; ++j) u.h[j] = (__bf16)fbuf[j];
    uint4* dstA = reinterpret_cast<uint4*>(&lA[buf][srow * 32 + koff]);
    dstA[0] = u.q[0];
    dstA[1] = u.q[1];
    uint4* dstB = reinterpret_cast<uint4*>(&lB[buf][srow * 32 + koff]);
    dstB[0] = rb[0];
    dstB[1] = rb[1];
  };

  load_regs(0);
  store_regs(0);
  __syncthreads();

  for (int kb = 0; kb < DM_; kb += 32) {
    int buf = (kb >> 5) & 1;
    bool more = (kb + 32) < DM_;
    if (more) load_regs(kb + 32);
    v16bf af[2], bf[4];
#pragma unroll
    for (int j = 0; j < 4; ++j) bf[j] = load_frag_b(lB[buf], 32, wn * 64 + j * 16, 0, lane);
#pragma unroll
    for (int i = 0; i < 2; ++i) af[i] = load_frag_a(lA[buf], 32, wm * 32 + i * 16, 0, lane);
#pragma unroll
    for (int i = 0; i < 2; ++i)
#pragma unroll
      for (int j = 0; j < 4; ++j) acc[i][j] = wmma_bf16(af[i], bf[j], acc[i][j]);
    if (more) store_regs(buf ^ 1);
    __syncthreads();
  }

#pragma unroll
  for (int i = 0; i < 2; ++i)
#pragma unroll
    for (int j = 0; j < 4; ++j)
#pragma unroll
      for (int r = 0; r < 8; ++r) {
        int m = m0 + wm * 32 + i * 16 + r + ((lane & 16) >> 1);
        int n = n0b + wn * 64 + j * 16 + (lane & 15);
        int b = m >> 10, s = m & 1023, h = n >> 6, d = n & 63;
        Out[(((size_t)h * B_ + b) * S_ + s) * DK_ + d] = (__bf16)acc[i][j][r];
      }
}

// ---------------------------------------------------------------- flash attention
// qh/kh/vh: bf16 [h][b][s][dk]; c: bf16 [b][s][h*64+d]
// 256 threads (8 waves), each wave owns 16 q rows; 64-key blocks,
// double-buffered K/Vt tiles with SW-pipelined loads.
__global__ __launch_bounds__(256) void mha_attn(
    const __bf16* __restrict__ qh, const __bf16* __restrict__ kh,
    const __bf16* __restrict__ vh, __bf16* __restrict__ cbuf) {
  __shared__ __align__(16) __bf16 lK[2][64 * 64];   // [key][d]
  __shared__ __align__(16) __bf16 lVt[2][64 * 64];  // [d][key]
  __shared__ __align__(16) __bf16 lP[8][16 * 64];
  int tid = threadIdx.x, lane = tid & 31, w = tid >> 5;
  int qb = blockIdx.x * 128;
  int hb = blockIdx.y;  // h*8 + b
  size_t base = (size_t)hb * S_ * DK_;
  const __bf16* Q = qh + base;
  const __bf16* K = kh + base;
  const __bf16* V = vh + base;
  int qr0 = qb + w * 16;

  v16bf qf0 = load_frag_a(Q, DK_, qr0, 0, lane);
  v16bf qf1 = load_frag_a(Q, DK_, qr0, 32, lane);
  v8f oacc[4] = {};
  float m_i[8], l_i[8];
#pragma unroll
  for (int r = 0; r < 8; ++r) { m_i[r] = -3.0e38f; l_i[r] = 0.0f; }
  const float scale = 0.125f;  // 1/sqrt(64)

  int krow = tid >> 2;           // 0..63
  int kcol = (tid & 3) * 16;     // 0,16,32,48
  uint4 rk[2], rv[2];
  auto load_regs = [&](int kb) {
    const uint4* srcK = reinterpret_cast<const uint4*>(K + (size_t)(kb + krow) * DK_ + kcol);
    rk[0] = srcK[0]; rk[1] = srcK[1];
    const uint4* srcV = reinterpret_cast<const uint4*>(V + (size_t)(kb + krow) * DK_ + kcol);
    rv[0] = srcV[0]; rv[1] = srcV[1];
    if (kb + 64 < S_) {
      __builtin_prefetch(srcK + 64 * DK_ / 8, 0, 1);
      __builtin_prefetch(srcV + 64 * DK_ / 8, 0, 1);
    }
  };
  auto store_regs = [&](int buf) {
    uint4* dstK = reinterpret_cast<uint4*>(&lK[buf][krow * 64 + kcol]);
    dstK[0] = rk[0];
    dstK[1] = rk[1];
    union { uint4 q[2]; __bf16 h[16]; } u;
    u.q[0] = rv[0];
    u.q[1] = rv[1];
#pragma unroll
    for (int d2 = 0; d2 < 16; ++d2) lVt[buf][(kcol + d2) * 64 + krow] = u.h[d2];
  };

  load_regs(0);
  store_regs(0);
  __syncthreads();

  for (int kb = 0; kb < S_; kb += 64) {
    int buf = (kb >> 6) & 1;
    bool more = (kb + 64) < S_;
    if (more) load_regs(kb + 64);

    // scores: 16 q rows x 64 keys
    v8f sacc[4];
#pragma unroll
    for (int nt = 0; nt < 4; ++nt) {
      v8f z = {0, 0, 0, 0, 0, 0, 0, 0};
      v16bf bk0 = load_frag_b(lK[buf], 64, nt * 16, 0, lane);
      v16bf bk1 = load_frag_b(lK[buf], 64, nt * 16, 32, lane);
      z = wmma_bf16(qf0, bk0, z);
      z = wmma_bf16(qf1, bk1, z);
      sacc[nt] = z;
    }

    // online softmax (rows live per-VGPR; reduce across 16-lane half-groups)
#pragma unroll
    for (int r = 0; r < 8; ++r) {
      float rm = fmaxf(fmaxf(sacc[0][r], sacc[1][r]), fmaxf(sacc[2][r], sacc[3][r]));
      rm = fmaxf(rm, __shfl_xor(rm, 1));
      rm = fmaxf(rm, __shfl_xor(rm, 2));
      rm = fmaxf(rm, __shfl_xor(rm, 4));
      rm = fmaxf(rm, __shfl_xor(rm, 8));
      float mnew = fmaxf(m_i[r], rm * scale);
      float alpha = __expf(m_i[r] - mnew);
      float rs = 0.0f;
#pragma unroll
      for (int nt = 0; nt < 4; ++nt) {
        float p = __expf(sacc[nt][r] * scale - mnew);
        sacc[nt][r] = p;
        rs += p;
      }
      rs += __shfl_xor(rs, 1);
      rs += __shfl_xor(rs, 2);
      rs += __shfl_xor(rs, 4);
      rs += __shfl_xor(rs, 8);
      l_i[r] = l_i[r] * alpha + rs;
      m_i[r] = mnew;
#pragma unroll
      for (int dt = 0; dt < 4; ++dt) oacc[dt][r] *= alpha;
      int prow = r + ((lane & 16) >> 1);
#pragma unroll
      for (int nt = 0; nt < 4; ++nt)
        lP[w][prow * 64 + nt * 16 + (lane & 15)] = (__bf16)sacc[nt][r];
    }

    // P x V (per-wave LDS round-trip; same-wave DS ops are in-order)
    v16bf pf0 = load_frag_a(&lP[w][0], 64, 0, 0, lane);
    v16bf pf1 = load_frag_a(&lP[w][0], 64, 0, 32, lane);
#pragma unroll
    for (int dt = 0; dt < 4; ++dt) {
      v16bf bv0 = load_frag_b(lVt[buf], 64, dt * 16, 0, lane);
      v16bf bv1 = load_frag_b(lVt[buf], 64, dt * 16, 32, lane);
      oacc[dt] = wmma_bf16(pf0, bv0, oacc[dt]);
      oacc[dt] = wmma_bf16(pf1, bv1, oacc[dt]);
    }

    if (more) store_regs(buf ^ 1);
    __syncthreads();
  }

  // epilogue: normalize and write c[b][s][h*64+d] bf16
  int h = hb >> 3, b = hb & 7;
#pragma unroll
  for (int r = 0; r < 8; ++r) {
    float inv = 1.0f / l_i[r];
    int s = qr0 + r + ((lane & 16) >> 1);
#pragma unroll
    for (int dt = 0; dt < 4; ++dt) {
      int d = dt * 16 + (lane & 15);
      cbuf[((size_t)(b * S_ + s)) * DM_ + h * DK_ + d] = (__bf16)(oacc[dt][r] * inv);
    }
  }
}

// ---------------------------------------------------------------- output GEMM
// A: c bf16 [8192 x 1024], Bt: Wo bf16 [N][K], out f32 + bias.
// 256 threads (8 waves), WG tile 128x128, wave tile 32x64, double-buffered LDS.
__global__ __launch_bounds__(256) void mha_out_gemm(
    const __bf16* __restrict__ A, const __bf16* __restrict__ Bt,
    const float* __restrict__ bias, float* __restrict__ Out) {
  __shared__ __align__(16) __bf16 lA[2][128 * 32];
  __shared__ __align__(16) __bf16 lB[2][128 * 32];
  int tid = threadIdx.x, lane = tid & 31, w = tid >> 5;
  int m0 = blockIdx.x * 128, n0b = blockIdx.y * 128;
  int wm = w >> 1, wn = w & 1;
  v8f acc[2][4] = {};

  int srow = tid >> 1;
  int koff = (tid & 1) * 16;
  uint4 raq[2], rbq[2];
  auto load_regs = [&](int kb) {
    const uint4* srcA = reinterpret_cast<const uint4*>(A + (size_t)(m0 + srow) * DM_ + kb + koff);
    raq[0] = srcA[0]; raq[1] = srcA[1];
    const uint4* srcB = reinterpret_cast<const uint4*>(Bt + (size_t)(n0b + srow) * DM_ + kb + koff);
    rbq[0] = srcB[0]; rbq[1] = srcB[1];
    if (kb + 32 < DM_) {
      __builtin_prefetch(srcA + 4, 0, 1);
      __builtin_prefetch(srcB + 4, 0, 1);
    }
  };
  auto store_regs = [&](int buf) {
    uint4* dstA = reinterpret_cast<uint4*>(&lA[buf][srow * 32 + koff]);
    dstA[0] = raq[0];
    dstA[1] = raq[1];
    uint4* dstB = reinterpret_cast<uint4*>(&lB[buf][srow * 32 + koff]);
    dstB[0] = rbq[0];
    dstB[1] = rbq[1];
  };

  load_regs(0);
  store_regs(0);
  __syncthreads();

  for (int kb = 0; kb < DM_; kb += 32) {
    int buf = (kb >> 5) & 1;
    bool more = (kb + 32) < DM_;
    if (more) load_regs(kb + 32);
    v16bf af[2], bf[4];
#pragma unroll
    for (int j = 0; j < 4; ++j) bf[j] = load_frag_b(lB[buf], 32, wn * 64 + j * 16, 0, lane);
#pragma unroll
    for (int i = 0; i < 2; ++i) af[i] = load_frag_a(lA[buf], 32, wm * 32 + i * 16, 0, lane);
#pragma unroll
    for (int i = 0; i < 2; ++i)
#pragma unroll
      for (int j = 0; j < 4; ++j) acc[i][j] = wmma_bf16(af[i], bf[j], acc[i][j]);
    if (more) store_regs(buf ^ 1);
    __syncthreads();
  }

#pragma unroll
  for (int i = 0; i < 2; ++i)
#pragma unroll
    for (int j = 0; j < 4; ++j)
#pragma unroll
      for (int r = 0; r < 8; ++r) {
        int m = m0 + wm * 32 + i * 16 + r + ((lane & 16) >> 1);
        int n = n0b + wn * 64 + j * 16 + (lane & 15);
        Out[(size_t)m * DM_ + n] = acc[i][j][r] + bias[n];
      }
}

extern "C" void kernel_launch(void* const* d_in, const int* in_sizes, int n_in,
                              void* d_out, int out_size, void* d_ws, size_t ws_size,
                              hipStream_t stream) {
  const float* q  = (const float*)d_in[0];
  const float* k  = (const float*)d_in[1];
  const float* v  = (const float*)d_in[2];
  const float* Wq = (const float*)d_in[3];
  const float* Wk = (const float*)d_in[4];
  const float* Wv = (const float*)d_in[5];
  const float* Wo = (const float*)d_in[6];
  const float* bo = (const float*)d_in[7];
  float* out = (float*)d_out;

  char* ws = (char*)d_ws;
  const size_t HBSD = (size_t)H_ * B_ * S_ * DK_;  // 8.39M elems
  __bf16* qh  = (__bf16*)(ws);
  __bf16* kh  = (__bf16*)(ws + 2 * HBSD);
  __bf16* vh  = (__bf16*)(ws + 4 * HBSD);
  __bf16* cbf = (__bf16*)(ws + 6 * HBSD);
  __bf16* WtQ = (__bf16*)(ws + 8 * HBSD);
  __bf16* WtK = (__bf16*)(ws + 8 * HBSD + 2u * DM_ * DM_);
  __bf16* WtV = (__bf16*)(ws + 8 * HBSD + 4u * DM_ * DM_);
  __bf16* WoB = (__bf16*)(ws + 8 * HBSD + 6u * DM_ * DM_);

  mha_prep_weights<<<(DM_ * DM_) / 256, 256, 0, stream>>>(Wq, Wk, Wv, Wo, WtQ, WtK, WtV, WoB);

  dim3 ggrid(B_ * S_ / 128, DM_ / 128);  // 64 x 8
  mha_proj_gemm<<<ggrid, 256, 0, stream>>>(q, WtQ, qh);
  mha_proj_gemm<<<ggrid, 256, 0, stream>>>(k, WtK, kh);
  mha_proj_gemm<<<ggrid, 256, 0, stream>>>(v, WtV, vh);

  dim3 agrid(S_ / 128, H_ * B_);  // 8 x 128
  mha_attn<<<agrid, 256, 0, stream>>>(qh, kh, vh, cbf);

  mha_out_gemm<<<ggrid, 256, 0, stream>>>(cbf, WoB, bo, out);
}
